// GraphViT_28458453303852
// MI455X (gfx1250) — matrix operand order repaired and checked by reading, not compiled, not run on hardware
//
#include <hip/hip_runtime.h>
#include <hip/hip_bf16.h>
#include <math.h>

// ---------------- problem constants ----------------
#define NN   16384   // nodes
#define CC   128     // clusters
#define KCL  128     // nodes per cluster
#define MM   98304   // edges
#define POSD 64      // node positional dim
#define WSZ  512     // W_SIZE
#define EDIM 544     // attention embed dim
#define HD   136     // head dim
#define NH   4

typedef __attribute__((ext_vector_type(16))) _Float16 v16h;
typedef __attribute__((ext_vector_type(8)))  float    v8f;

// ---------------- generic WMMA GEMM ----------------
// C[M,N] = act( scale * (A @ W) + bias ) + res
// A rows are assembled on the fly from up to 8 gathered segments.
struct Seg {
  const float* p;       // source matrix
  int ld;               // row stride of source
  int w;                // columns contributed
  const int* idx;       // row indirection (nullptr = identity)
  int idxStride, idxOff;
  const int* idx2;      // second-level indirection (nullptr = none)
};

struct GemmArgs {
  Seg seg[8]; int nseg;
  const float* W; int ldw; int transB;   // W[k*ldw+n] or W[n*ldw+k]
  const float* bias;
  const float* res; int ldres;
  float* C; int ldc;
  float scale; int act;                  // 0 none, 1 relu, 2 tanh
  int M, N, K;
};

// Per-k-column segment resolution (hoisted out of the row loop: each thread's
// k-column is fixed, so this runs once per k-tile, not once per element).
struct SegRes {
  const float* p;       // pre-offset by column
  int ld;
  const int* idx; int idxStride, idxOff;
  const int* idx2;
  int valid;
};

__device__ __forceinline__ SegRes resolveSeg(const GemmArgs& g, int k) {
  SegRes r; r.valid = 0; r.p = nullptr; r.ld = 0;
  r.idx = nullptr; r.idxStride = 0; r.idxOff = 0; r.idx2 = nullptr;
  if (k < g.K) {
    int off = k;
    for (int s = 0; s < g.nseg; ++s) {
      const Seg& S = g.seg[s];
      if (off < S.w) {
        r.p = S.p + off; r.ld = S.ld;
        r.idx = S.idx; r.idxStride = S.idxStride; r.idxOff = S.idxOff;
        r.idx2 = S.idx2; r.valid = 1;
        break;
      }
      off -= S.w;
    }
  }
  return r;
}

__device__ __forceinline__ float loadSeg(const SegRes& r, int row) {
  if (!r.valid) return 0.f;
  int rr = row;
  if (r.idx)  rr = r.idx[(long long)row * r.idxStride + r.idxOff];
  if (r.idx2) rr = r.idx2[rr];
  return r.p[(long long)rr * r.ld];
}

__global__ __launch_bounds__(256) void k_gemm(GemmArgs g) {
  __shared__ _Float16 As[32][72];   // [m][k], 64-wide k-tile + pad (144B rows)
  __shared__ _Float16 Bs[64][72];   // [n][k] transposed staging
  int tid = threadIdx.x, lane = tid & 31, wave = tid >> 5;
  int m0 = blockIdx.y * 32, n0 = blockIdx.x * 64;
  int wm = (wave >> 2) * 16, wn = (wave & 3) * 16;
  union { v8f v; float e[8]; } acc;
#pragma unroll
  for (int r = 0; r < 8; r++) acc.e[r] = 0.f;

  int kk  = tid & 31;     // fixed k-column pair for this thread
  int ri0 = tid >> 5;     // starting row for staging loops

  int Kt = (g.K + 63) >> 6;
  for (int kt = 0; kt < Kt; kt++) {
    int k0 = kt * 64;

    // ---- prefetch next weight tile into L2 (global_prefetch_b8) ----
    if (kt + 1 < Kt) {
      int nk = k0 + 64 + ri0 * 8;
      if (nk < g.K) {
        const float* pf = g.transB ? &g.W[(long long)(n0 + kk) * g.ldw + nk]
                                   : &g.W[(long long)nk * g.ldw + n0 + kk];
        __builtin_prefetch((const void*)pf, 0, 1);
      }
    }

    // ---- stage A: resolve the two k-columns once, then sweep rows ----
    SegRes s0 = resolveSeg(g, k0 + kk);
    SegRes s1 = resolveSeg(g, k0 + kk + 32);
    for (int mi = ri0; mi < 32; mi += 8) {
      int gr = m0 + mi;
      float v0 = 0.f, v1 = 0.f;
      if (gr < g.M) { v0 = loadSeg(s0, gr); v1 = loadSeg(s1, gr); }
      As[mi][kk]      = (_Float16)v0;
      As[mi][kk + 32] = (_Float16)v1;
    }
    // ---- stage B (transposed) ----
    int gk0 = k0 + kk, gk1 = gk0 + 32;
    for (int ni = ri0; ni < 64; ni += 8) {
      int gn = n0 + ni;
      float v0 = 0.f, v1 = 0.f;
      if (gn < g.N) {
        if (gk0 < g.K) v0 = g.transB ? g.W[(long long)gn * g.ldw + gk0]
                                     : g.W[(long long)gk0 * g.ldw + gn];
        if (gk1 < g.K) v1 = g.transB ? g.W[(long long)gn * g.ldw + gk1]
                                     : g.W[(long long)gk1 * g.ldw + gn];
      }
      Bs[ni][kk]      = (_Float16)v0;
      Bs[ni][kk + 32] = (_Float16)v1;
    }
    __syncthreads();

    int m  = wm + (lane & 15);
    int ab = (lane & 16) ? 8 : 0;
    int n  = wn + (lane & 15);
    int bb = (lane & 16) ? 8 : 0;
#pragma unroll
    for (int hlf = 0; hlf < 2; hlf++) {
      int kb = hlf * 32;
      union { v16h v; _Float16 e[16]; } a, b;
#pragma unroll
      for (int h = 0; h < 16; h++) a.e[h] = As[m][kb + (h < 8 ? h : h + 8) + ab];
#pragma unroll
      for (int h = 0; h < 16; h++) b.e[h] = Bs[n][kb + bb + (h < 8 ? h : h + 8)];
      acc.v = __builtin_amdgcn_wmma_f32_16x16x32_f16(
          false, a.v, false, b.v, (short)0, acc.v, false, false);
    }
    __syncthreads();
  }

  int n = n0 + wn + (lane & 15);
  if (n < g.N) {
#pragma unroll
    for (int r = 0; r < 8; r++) {
      int m = m0 + wm + r + ((lane & 16) ? 8 : 0);
      if (m < g.M) {
        float v = acc.e[r] * g.scale;
        if (g.bias) v += g.bias[n];
        if (g.act == 1) v = fmaxf(v, 0.f);
        else if (g.act == 2) v = tanhf(v);
        if (g.res) v += g.res[(long long)m * g.ldres + n];
        g.C[(long long)m * g.ldc + n] = v;
      }
    }
  }
}

// ---------------- positional encodings ----------------
__device__ __forceinline__ void embed2(float px, float py, float* o) {
#pragma unroll
  for (int j = 0; j < 8; j++) {
    float f = 3.14159265358979323846f * exp2f((float)j - 3.f);
    o[j]      = cosf(px * f);
    o[8 + j]  = sinf(px * f);
    o[16 + j] = cosf(py * f);
    o[24 + j] = sinf(py * f);
  }
}

__global__ void k_centers(const float* mp, const int* clusters,
                          const unsigned char* cmask, float* centers,
                          float* posclu, int* cidx) {
  __shared__ float sx[128], sy[128];
  __shared__ int sm[128];
  __shared__ float cxy[2];
  int c = blockIdx.x, t = threadIdx.x;
  int node = clusters[c * KCL + t];
  sx[t] = mp[node * 2]; sy[t] = mp[node * 2 + 1];
  sm[t] = cmask[c * KCL + t] ? 1 : 0;
  __syncthreads();
  for (int s = 64; s > 0; s >>= 1) {
    if (t < s) { sx[t] += sx[t + s]; sy[t] += sy[t + s]; sm[t] += sm[t + s]; }
    __syncthreads();
  }
  if (t == 0) {
    float d = (float)sm[0] + 1e-8f;
    cxy[0] = sx[0] / d; cxy[1] = sy[0] / d;
    centers[c * 2] = cxy[0]; centers[c * 2 + 1] = cxy[1];
    int id = sm[0] - 1; if (id == -1) id = KCL - 1;
    cidx[c] = id;
  }
  __syncthreads();
  if (t == 0) embed2(cxy[0], cxy[1], posclu + c * 32);
}

__global__ void k_posnode(const float* mp, const int* clusters,
                          const float* centers, float* posnode, int* n2c) {
  int i = blockIdx.x * blockDim.x + threadIdx.x;
  if (i >= CC * KCL) return;
  int c = i / KCL;
  int node = clusters[i];
  n2c[node] = c;
  float px = mp[node * 2], py = mp[node * 2 + 1];
  float* o = posnode + (long long)node * POSD;
  embed2(px, py, o);
  embed2(centers[c * 2] - px, centers[c * 2 + 1] - py, o + 32);
}

__global__ void k_edgefeat(const float* mp, const int* edges, float* ein) {
  int e = blockIdx.x * blockDim.x + threadIdx.x;
  if (e >= MM) return;
  int s = edges[e * 2], r = edges[e * 2 + 1];
  float dx = mp[s * 2] - mp[r * 2];
  float dy = mp[s * 2 + 1] - mp[r * 2 + 1];
  ein[e * 3] = dx; ein[e * 3 + 1] = dy;
  ein[e * 3 + 2] = sqrtf(dx * dx + dy * dy);
}

// ---------------- elementwise / reduction helpers ----------------
__global__ void k_zero(float* p, long long n) {
  long long i = (long long)blockIdx.x * blockDim.x + threadIdx.x;
  if (i < n) p[i] = 0.f;
}
__global__ void k_add(float* y, const float* x, long long n) {
  long long i = (long long)blockIdx.x * blockDim.x + threadIdx.x;
  if (i < n) y[i] += x[i];
}
__global__ void k_scatter(const float* ef, const int* edges, float* agg) {
  long long i = (long long)blockIdx.x * blockDim.x + threadIdx.x;
  if (i >= (long long)MM * 128) return;
  int e = (int)(i >> 7), c = (int)(i & 127);
  atomicAdd(&agg[(long long)edges[e * 2] * 128 + c], ef[i]);
}

// out[row] = res + LN(x[row]) * g + b
__global__ void k_ln(const float* x, int ldx, const float* g, const float* b,
                     const float* res, int ldres, float* o, int ldo, int D) {
  __shared__ float red[128];
  int row = blockIdx.x, t = threadIdx.x;
  float loc[4];
  int nv = (D + 127) / 128;
  float s = 0.f;
  for (int i = 0; i < nv; i++) {
    int c = t + i * 128;
    float v = (c < D) ? x[(long long)row * ldx + c] : 0.f;
    loc[i] = v; s += v;
  }
  red[t] = s; __syncthreads();
  for (int st = 64; st > 0; st >>= 1) { if (t < st) red[t] += red[t + st]; __syncthreads(); }
  float mean = red[0] / (float)D; __syncthreads();
  s = 0.f;
  for (int i = 0; i < nv; i++) {
    int c = t + i * 128;
    if (c < D) { float d = loc[i] - mean; s += d * d; }
  }
  red[t] = s; __syncthreads();
  for (int st = 64; st > 0; st >>= 1) { if (t < st) red[t] += red[t + st]; __syncthreads(); }
  float rstd = rsqrtf(red[0] / (float)D + 1e-5f);
  for (int i = 0; i < nv; i++) {
    int c = t + i * 128;
    if (c < D) {
      float v = (loc[i] - mean) * rstd * g[c] + b[c];
      if (res) v += res[(long long)row * ldres + c];
      o[(long long)row * ldo + c] = v;
    }
  }
}

__global__ void k_softmax(float* sc) {   // rows of length 128 (mask pre-added)
  __shared__ float red[128];
  int row = blockIdx.x, t = threadIdx.x;
  float v = sc[(long long)row * 128 + t];
  red[t] = v; __syncthreads();
  for (int s = 64; s > 0; s >>= 1) { if (t < s) red[t] = fmaxf(red[t], red[t + s]); __syncthreads(); }
  float mx = red[0]; __syncthreads();
  float e = __expf(v - mx);
  red[t] = e; __syncthreads();
  for (int s = 64; s > 0; s >>= 1) { if (t < s) red[t] += red[t + s]; __syncthreads(); }
  sc[(long long)row * 128 + t] = e / red[0];
}

// ---------------- GRU ----------------
__global__ void k_cvt_whT(const float* wh, _Float16* whT) {
  int i = blockIdx.x * blockDim.x + threadIdx.x;
  if (i >= 1536 * 512) return;
  int n = i / 512, k = i % 512;
  whT[i] = (_Float16)wh[(long long)k * 1536 + n];
}

// 8 blocks x 16 clusters each; h in LDS; gh = h @ wh via WMMA against whT (f16)
__global__ __launch_bounds__(256) void k_gru(const float* xi, const _Float16* whT,
                                             const float* bh, const int* cidx,
                                             float* ghbuf, float* Wlast) {
  __shared__ float    hf[16][512];
  __shared__ _Float16 hh[16][520];
  int tid = threadIdx.x, lane = tid & 31, wave = tid >> 5;
  int c0 = blockIdx.x * 16;
  for (int i = tid; i < 16 * 512; i += 256) {
    hf[i >> 9][i & 511] = 0.f;
    hh[i >> 9][i & 511] = (_Float16)0.f;
  }
  __syncthreads();

  for (int t = 0; t < KCL; t++) {
    // gh = h @ wh + bh   (M=16 clusters, N=1536, K=512)
    for (int nt = wave; nt < 96; nt += 8) {
      union { v8f v; float e[8]; } acc;
#pragma unroll
      for (int r = 0; r < 8; r++) acc.e[r] = 0.f;
      int m  = lane & 15;
      int ab = (lane & 16) ? 8 : 0;
      int bb = (lane & 16) ? 8 : 0;
      int n  = nt * 16 + (lane & 15);
      for (int kt = 0; kt < 16; kt++) {
        int k0 = kt * 32;
        union { v16h v; _Float16 e[16]; } a, b;
#pragma unroll
        for (int h = 0; h < 16; h++) a.e[h] = hh[m][k0 + (h < 8 ? h : h + 8) + ab];
        const _Float16* bp = whT + (long long)n * 512 + k0 + bb;
#pragma unroll
        for (int h = 0; h < 16; h++) b.e[h] = bp[(h < 8 ? h : h + 8)];
        acc.v = __builtin_amdgcn_wmma_f32_16x16x32_f16(
            false, a.v, false, b.v, (short)0, acc.v, false, false);
      }
#pragma unroll
      for (int r = 0; r < 8; r++) {
        int mm = r + ((lane & 16) ? 8 : 0);
        ghbuf[(long long)(c0 + mm) * 1536 + n] = acc.e[r] + bh[n];
      }
    }
    __threadfence_block();
    __syncthreads();
    // gates (torch order r,z,n)
    for (int i = tid; i < 16 * 512; i += 256) {
      int ci = i >> 9, j = i & 511;
      int c = c0 + ci;
      long long xrow = ((long long)c * KCL + t) * 1536;
      const float* gh = ghbuf + (long long)c * 1536;
      float r = 1.f / (1.f + __expf(-(xi[xrow + j] + gh[j])));
      float z = 1.f / (1.f + __expf(-(xi[xrow + 512 + j] + gh[512 + j])));
      float nn = tanhf(xi[xrow + 1024 + j] + r * gh[1024 + j]);
      float hn = (1.f - z) * nn + z * hf[ci][j];
      hf[ci][j] = hn;
      if (t == cidx[c]) Wlast[(long long)c * 512 + j] = hn;
    }
    __syncthreads();
    for (int i = tid; i < 16 * 512; i += 256)
      hh[i >> 9][i & 511] = (_Float16)hf[i >> 9][i & 511];
    __syncthreads();
  }
}

// ---------------- host orchestration ----------------
struct LinP { const float *b, *w; };
struct LNP  { const float *b, *g; };

static inline void launch_gemm(hipStream_t s, const GemmArgs& g) {
  dim3 grid((unsigned)((g.N + 63) / 64), (unsigned)((g.M + 31) / 32), 1);
  k_gemm<<<grid, 256, 0, s>>>(g);
}
static inline GemmArgs mk() { GemmArgs g{}; g.scale = 1.f; return g; }
static inline Seg sid(const float* p, int ld, int w) { return Seg{p, ld, w, nullptr, 0, 0, nullptr}; }
static inline Seg sgx(const float* p, int ld, int w, const int* idx, int st, int off,
                      const int* idx2 = nullptr) { return Seg{p, ld, w, idx, st, off, idx2}; }

extern "C" void kernel_launch(void* const* d_in, const int* in_sizes, int n_in,
                              void* d_out, int out_size, void* d_ws, size_t ws_size,
                              hipStream_t stream) {
  const float* mesh_pos  = (const float*)d_in[0];
  const float* states    = (const float*)d_in[1];
  const float* node_type = (const float*)d_in[2];
  const float* amask     = (const float*)d_in[3];
  const int*   edges     = (const int*)d_in[4];
  const int*   clusters  = (const int*)d_in[5];
  const unsigned char* cmask = (const unsigned char*)d_in[6];

  // ---- params: jax tree-flatten order (sorted dict keys, list order) ----
  int ip = 7;
  auto P = [&]() -> const float* { return (const float*)d_in[ip++]; };
  struct AttnP { const float *in_b, *in_w; LinP lin; LNP ln1, ln2;
                 LinP ml0, ml1; const float *out_b, *out_w; } at[4];
  for (int i = 0; i < 4; i++) {             // "attn"
    at[i].in_b = P(); at[i].in_w = P();
    at[i].lin.b = P(); at[i].lin.w = P();
    at[i].ln1.b = P(); at[i].ln1.g = P();
    at[i].ln2.b = P(); at[i].ln2.g = P();
    at[i].ml0.b = P(); at[i].ml0.w = P();
    at[i].ml1.b = P(); at[i].ml1.w = P();
    at[i].out_b = P(); at[i].out_w = P();
  }
  LinP ee0{P(), P()}, ee1{P(), P()};        // encoder.edge
  struct MlpLN { LinP l0, l1, l2; LNP ln; } ge[4], gn[4];
  for (int i = 0; i < 4; i++) {             // encoder.gns
    ge[i].l0 = {P(), P()}; ge[i].l1 = {P(), P()}; ge[i].l2 = {P(), P()};
    ge[i].ln = {P(), P()};
    gn[i].l0 = {P(), P()}; gn[i].l1 = {P(), P()}; gn[i].l2 = {P(), P()};
    gn[i].ln = {P(), P()};
  }
  LinP en0{P(), P()}, en1{P(), P()};        // encoder.node
  LNP lnf{P(), P()};                        // ln
  const float *g_bh = P(), *g_bi = P(), *g_wh = P(), *g_wi = P(); // pooling.gru
  LinP lr0{P(), P()}, lr1{P(), P()};        // pooling.linear_rnn
  LinP fm0{P(), P()}, fm1{P(), P()}, fm2{P(), P()};               // retrieve.fm
  LinP re0{P(), P()}, re1{P(), P()}, re2{P(), P()};               // retrieve.gnn.f_edge
  LinP rn0{P(), P()}, rn1{P(), P()}, rn2{P(), P()};               // retrieve.gnn.f_node
  (void)n_in; (void)in_sizes; (void)out_size; (void)ws_size;

  // ---- workspace arena ----
  char* wp = (char*)d_ws;
  auto alloc = [&](size_t bytes) -> void* {
    void* r = wp; wp += (bytes + 255) & ~(size_t)255; return r;
  };
  float* posnode = (float*)alloc((size_t)NN * POSD * 4);
  float* posclu  = (float*)alloc((size_t)CC * 32 * 4);
  float* centers = (float*)alloc((size_t)CC * 2 * 4);
  int*   n2c     = (int*)alloc((size_t)NN * 4);
  int*   cidx    = (int*)alloc((size_t)CC * 4);
  float* V       = (float*)alloc((size_t)NN * 128 * 4);
  float* E       = (float*)alloc((size_t)MM * 128 * 4);
  float* big     = (float*)alloc((size_t)3 * MM * 128 * 4);  // te1/te2/eln, xi aliases
  float* te1 = big;
  float* te2 = big + (size_t)MM * 128;
  float* eln = big + (size_t)2 * MM * 128;
  float* xi  = big;                                          // [16384,1536] during pooling
  float* agg = (float*)alloc((size_t)NN * 128 * 4);
  float* nb1 = (float*)alloc((size_t)NN * 128 * 4);
  float* nb2 = (float*)alloc((size_t)NN * 128 * 4);
  float* nb3 = (float*)alloc((size_t)NN * 128 * 4);
  float* ein = (float*)alloc((size_t)MM * 3 * 4);
  float* ghb = (float*)alloc((size_t)CC * 1536 * 4);
  _Float16* whT = (_Float16*)alloc((size_t)1536 * 512 * 2);
  float* Wbuf = (float*)alloc((size_t)CC * WSZ * 4);
  float* Wl   = (float*)alloc((size_t)CC * WSZ * 4);
  float* W1   = (float*)alloc((size_t)CC * WSZ * 4);
  float* qkv  = (float*)alloc((size_t)CC * 3 * EDIM * 4);
  float* scor = (float*)alloc((size_t)NH * CC * CC * 4);
  float* ctx  = (float*)alloc((size_t)CC * EDIM * 4);
  float* W2   = (float*)alloc((size_t)CC * EDIM * 4);
  float* W3   = (float*)alloc((size_t)CC * WSZ * 4);
  float* ta1  = (float*)alloc((size_t)CC * WSZ * 4);
  float* ta2  = (float*)alloc((size_t)CC * WSZ * 4);
  float* tc1  = (float*)alloc((size_t)CC * 128 * 4);

  // ---- positional encodings ----
  k_centers <<<CC, 128, 0, stream>>>(mesh_pos, clusters, cmask, centers, posclu, cidx);
  k_posnode <<<(CC * KCL + 255) / 256, 256, 0, stream>>>(mesh_pos, clusters, centers, posnode, n2c);
  k_edgefeat<<<(MM + 255) / 256, 256, 0, stream>>>(mesh_pos, edges, ein);

  // ---- encoder: node MLP [13->128 relu ->128] ----
  { GemmArgs g = mk();
    g.seg[0] = sid(states, 4, 4); g.seg[1] = sid(node_type, 9, 9); g.nseg = 2;
    g.W = en0.w; g.ldw = 128; g.bias = en0.b; g.C = nb1; g.ldc = 128;
    g.act = 1; g.M = NN; g.N = 128; g.K = 13; launch_gemm(stream, g); }
  { GemmArgs g = mk();
    g.seg[0] = sid(nb1, 128, 128); g.nseg = 1;
    g.W = en1.w; g.ldw = 128; g.bias = en1.b; g.C = V; g.ldc = 128;
    g.M = NN; g.N = 128; g.K = 128; launch_gemm(stream, g); }
  // ---- encoder: edge MLP [3->128 relu ->128] ----
  { GemmArgs g = mk();
    g.seg[0] = sid(ein, 3, 3); g.nseg = 1;
    g.W = ee0.w; g.ldw = 128; g.bias = ee0.b; g.C = te1; g.ldc = 128;
    g.act = 1; g.M = MM; g.N = 128; g.K = 3; launch_gemm(stream, g); }
  { GemmArgs g = mk();
    g.seg[0] = sid(te1, 128, 128); g.nseg = 1;
    g.W = ee1.w; g.ldw = 128; g.bias = ee1.b; g.C = E; g.ldc = 128;
    g.M = MM; g.N = 128; g.K = 128; launch_gemm(stream, g); }

  // ---- 4 GNN rounds ----
  for (int r = 0; r < 4; r++) {
    // f_edge: concat(s[V,pos], r[V,pos], E) K=512
    { GemmArgs g = mk();
      g.seg[0] = sgx(V, 128, 128, edges, 2, 0);
      g.seg[1] = sgx(posnode, POSD, POSD, edges, 2, 0);
      g.seg[2] = sgx(V, 128, 128, edges, 2, 1);
      g.seg[3] = sgx(posnode, POSD, POSD, edges, 2, 1);
      g.seg[4] = sid(E, 128, 128); g.nseg = 5;
      g.W = ge[r].l0.w; g.ldw = 128; g.bias = ge[r].l0.b; g.C = te1; g.ldc = 128;
      g.act = 1; g.M = MM; g.N = 128; g.K = 512; launch_gemm(stream, g); }
    { GemmArgs g = mk(); g.seg[0] = sid(te1, 128, 128); g.nseg = 1;
      g.W = ge[r].l1.w; g.ldw = 128; g.bias = ge[r].l1.b; g.C = te2; g.ldc = 128;
      g.act = 1; g.M = MM; g.N = 128; g.K = 128; launch_gemm(stream, g); }
    { GemmArgs g = mk(); g.seg[0] = sid(te2, 128, 128); g.nseg = 1;
      g.W = ge[r].l2.w; g.ldw = 128; g.bias = ge[r].l2.b; g.C = te1; g.ldc = 128;
      g.M = MM; g.N = 128; g.K = 128; launch_gemm(stream, g); }
    k_ln<<<MM, 128, 0, stream>>>(te1, 128, ge[r].ln.g, ge[r].ln.b, nullptr, 0, eln, 128, 128);
    // agg = segsum(e, src)
    k_zero<<<(unsigned)(((long long)NN * 128 + 255) / 256), 256, 0, stream>>>(agg, (long long)NN * 128);
    k_scatter<<<(unsigned)(((long long)MM * 128 + 255) / 256), 256, 0, stream>>>(eln, edges, agg);
    // f_node: concat(V,pos,agg) K=320
    { GemmArgs g = mk();
      g.seg[0] = sid(V, 128, 128); g.seg[1] = sid(posnode, POSD, POSD);
      g.seg[2] = sid(agg, 128, 128); g.nseg = 3;
      g.W = gn[r].l0.w; g.ldw = 128; g.bias = gn[r].l0.b; g.C = nb1; g.ldc = 128;
      g.act = 1; g.M = NN; g.N = 128; g.K = 320; launch_gemm(stream, g); }
    { GemmArgs g = mk(); g.seg[0] = sid(nb1, 128, 128); g.nseg = 1;
      g.W = gn[r].l1.w; g.ldw = 128; g.bias = gn[r].l1.b; g.C = nb2; g.ldc = 128;
      g.act = 1; g.M = NN; g.N = 128; g.K = 128; launch_gemm(stream, g); }
    { GemmArgs g = mk(); g.seg[0] = sid(nb2, 128, 128); g.nseg = 1;
      g.W = gn[r].l2.w; g.ldw = 128; g.bias = gn[r].l2.b; g.C = nb1; g.ldc = 128;
      g.M = NN; g.N = 128; g.K = 128; launch_gemm(stream, g); }
    k_ln<<<NN, 128, 0, stream>>>(nb1, 128, gn[r].ln.g, gn[r].ln.b, nullptr, 0, nb2, 128, 128);
    k_add<<<(unsigned)(((long long)NN * 128 + 255) / 256), 256, 0, stream>>>(V, nb2, (long long)NN * 128);
    k_add<<<(unsigned)(((long long)MM * 128 + 255) / 256), 256, 0, stream>>>(E, eln, (long long)MM * 128);
  }

  // ---- pooling: xi = inp @ wi + bi  (rows = c*128+t, node = clusters[row]) ----
  { GemmArgs g = mk();
    g.seg[0] = sgx(V, 128, 128, clusters, 1, 0);
    g.seg[1] = sgx(posnode, POSD, POSD, clusters, 1, 0); g.nseg = 2;
    g.W = g_wi; g.ldw = 1536; g.bias = g_bi; g.C = xi; g.ldc = 1536;
    g.M = CC * KCL; g.N = 1536; g.K = 192; launch_gemm(stream, g); }
  k_cvt_whT<<<(1536 * 512 + 255) / 256, 256, 0, stream>>>(g_wh, whT);
  k_gru<<<8, 256, 0, stream>>>(xi, whT, g_bh, cidx, ghb, Wl);
  // linear_rnn: 512->128 relu ->512
  { GemmArgs g = mk(); g.seg[0] = sid(Wl, WSZ, WSZ); g.nseg = 1;
    g.W = lr0.w; g.ldw = 128; g.bias = lr0.b; g.C = tc1; g.ldc = 128;
    g.act = 1; g.M = CC; g.N = 128; g.K = WSZ; launch_gemm(stream, g); }
  { GemmArgs g = mk(); g.seg[0] = sid(tc1, 128, 128); g.nseg = 1;
    g.W = lr1.w; g.ldw = WSZ; g.bias = lr1.b; g.C = Wbuf; g.ldc = WSZ;
    g.M = CC; g.N = WSZ; g.K = 128; launch_gemm(stream, g); }

  // ---- 4 attention blocks (pre-LN) ----
  float iscale = 1.f / sqrtf((float)HD);
  for (int b = 0; b < 4; b++) {
    k_ln<<<CC, 128, 0, stream>>>(Wbuf, WSZ, at[b].ln1.g, at[b].ln1.b, nullptr, 0, W1, WSZ, WSZ);
    { GemmArgs g = mk();                                   // qkv = [W1,posclu] @ in_w + in_b
      g.seg[0] = sid(W1, WSZ, WSZ); g.seg[1] = sid(posclu, 32, 32); g.nseg = 2;
      g.W = at[b].in_w; g.ldw = 3 * EDIM; g.bias = at[b].in_b;
      g.C = qkv; g.ldc = 3 * EDIM; g.M = CC; g.N = 3 * EDIM; g.K = EDIM;
      launch_gemm(stream, g); }
    for (int h = 0; h < NH; h++) {                         // scores = q k^T / sqrt(hd) + mask
      GemmArgs g = mk();
      g.seg[0] = sid(qkv + h * HD, 3 * EDIM, HD); g.nseg = 1;
      g.W = qkv + EDIM + h * HD; g.ldw = 3 * EDIM; g.transB = 1;
      g.scale = iscale; g.res = amask; g.ldres = CC;
      g.C = scor + (size_t)h * CC * CC; g.ldc = CC;
      g.M = CC; g.N = CC; g.K = HD; launch_gemm(stream, g);
    }
    k_softmax<<<NH * CC, 128, 0, stream>>>(scor);
    for (int h = 0; h < NH; h++) {                         // ctx = P @ v
      GemmArgs g = mk();
      g.seg[0] = sid(scor + (size_t)h * CC * CC, CC, CC); g.nseg = 1;
      g.W = qkv + 2 * EDIM + h * HD; g.ldw = 3 * EDIM;
      g.C = ctx + h * HD; g.ldc = EDIM;
      g.M = CC; g.N = HD; g.K = CC; launch_gemm(stream, g);
    }
    { GemmArgs g = mk(); g.seg[0] = sid(ctx, EDIM, EDIM); g.nseg = 1;   // out proj
      g.W = at[b].out_w; g.ldw = EDIM; g.bias = at[b].out_b;
      g.C = W2; g.ldc = EDIM; g.M = CC; g.N = EDIM; g.K = EDIM; launch_gemm(stream, g); }
    { GemmArgs g = mk(); g.seg[0] = sid(W2, EDIM, EDIM); g.nseg = 1;    // W3 = W + linear(W2)
      g.W = at[b].lin.w; g.ldw = WSZ; g.bias = at[b].lin.b;
      g.res = Wbuf; g.ldres = WSZ; g.C = W3; g.ldc = WSZ;
      g.M = CC; g.N = WSZ; g.K = EDIM; launch_gemm(stream, g); }
    k_ln<<<CC, 128, 0, stream>>>(W3, WSZ, at[b].ln2.g, at[b].ln2.b, nullptr, 0, ta1, WSZ, WSZ);
    { GemmArgs g = mk(); g.seg[0] = sid(ta1, WSZ, WSZ); g.nseg = 1;     // mlp hidden
      g.W = at[b].ml0.w; g.ldw = WSZ; g.bias = at[b].ml0.b;
      g.C = ta2; g.ldc = WSZ; g.act = 1; g.M = CC; g.N = WSZ; g.K = WSZ; launch_gemm(stream, g); }
    { GemmArgs g = mk(); g.seg[0] = sid(ta2, WSZ, WSZ); g.nseg = 1;     // W = W3 + mlp out
      g.W = at[b].ml1.w; g.ldw = WSZ; g.bias = at[b].ml1.b;
      g.res = W3; g.ldres = WSZ; g.C = Wbuf; g.ldc = WSZ;
      g.M = CC; g.N = WSZ; g.K = WSZ; launch_gemm(stream, g); }
  }
  k_ln<<<CC, 128, 0, stream>>>(Wbuf, WSZ, lnf.g, lnf.b, nullptr, 0, Wbuf, WSZ, WSZ);

  // ---- retrieve GNN: nodes = [V, W[n2c], pos] (704) ----
  { GemmArgs g = mk();                                     // f_edge L0, K=1536
    g.seg[0] = sgx(V, 128, 128, edges, 2, 0);
    g.seg[1] = sgx(Wbuf, WSZ, WSZ, edges, 2, 0, n2c);
    g.seg[2] = sgx(posnode, POSD, POSD, edges, 2, 0);
    g.seg[3] = sgx(V, 128, 128, edges, 2, 1);
    g.seg[4] = sgx(Wbuf, WSZ, WSZ, edges, 2, 1, n2c);
    g.seg[5] = sgx(posnode, POSD, POSD, edges, 2, 1);
    g.seg[6] = sid(E, 128, 128); g.nseg = 7;
    g.W = re0.w; g.ldw = 128; g.bias = re0.b; g.C = te1; g.ldc = 128;
    g.act = 1; g.M = MM; g.N = 128; g.K = 1536; launch_gemm(stream, g); }
  { GemmArgs g = mk(); g.seg[0] = sid(te1, 128, 128); g.nseg = 1;
    g.W = re1.w; g.ldw = 128; g.bias = re1.b; g.C = te2; g.ldc = 128;
    g.act = 1; g.M = MM; g.N = 128; g.K = 128; launch_gemm(stream, g); }
  { GemmArgs g = mk(); g.seg[0] = sid(te2, 128, 128); g.nseg = 1;
    g.W = re2.w; g.ldw = 128; g.bias = re2.b; g.C = eln; g.ldc = 128;
    g.M = MM; g.N = 128; g.K = 128; launch_gemm(stream, g); }
  k_zero<<<(unsigned)(((long long)NN * 128 + 255) / 256), 256, 0, stream>>>(agg, (long long)NN * 128);
  k_scatter<<<(unsigned)(((long long)MM * 128 + 255) / 256), 256, 0, stream>>>(eln, edges, agg);
  { GemmArgs g = mk();                                     // f_node L0, K=832
    g.seg[0] = sid(V, 128, 128);
    g.seg[1] = sgx(Wbuf, WSZ, WSZ, n2c, 1, 0);
    g.seg[2] = sid(posnode, POSD, POSD);
    g.seg[3] = sid(agg, 128, 128); g.nseg = 4;
    g.W = rn0.w; g.ldw = 128; g.bias = rn0.b; g.C = nb1; g.ldc = 128;
    g.act = 1; g.M = NN; g.N = 128; g.K = 832; launch_gemm(stream, g); }
  { GemmArgs g = mk(); g.seg[0] = sid(nb1, 128, 128); g.nseg = 1;
    g.W = rn1.w; g.ldw = 128; g.bias = rn1.b; g.C = nb2; g.ldc = 128;
    g.act = 1; g.M = NN; g.N = 128; g.K = 128; launch_gemm(stream, g); }
  { GemmArgs g = mk(); g.seg[0] = sid(nb2, 128, 128); g.nseg = 1;
    g.W = rn2.w; g.ldw = 128; g.bias = rn2.b; g.C = nb1; g.ldc = 128;
    g.M = NN; g.N = 128; g.K = 128; launch_gemm(stream, g); }
  // fm: tanh, tanh, linear -> out = states + delta
  { GemmArgs g = mk(); g.seg[0] = sid(nb1, 128, 128); g.nseg = 1;
    g.W = fm0.w; g.ldw = 128; g.bias = fm0.b; g.C = nb2; g.ldc = 128;
    g.act = 2; g.M = NN; g.N = 128; g.K = 128; launch_gemm(stream, g); }
  { GemmArgs g = mk(); g.seg[0] = sid(nb2, 128, 128); g.nseg = 1;
    g.W = fm1.w; g.ldw = 128; g.bias = fm1.b; g.C = nb3; g.ldc = 128;
    g.act = 2; g.M = NN; g.N = 128; g.K = 128; launch_gemm(stream, g); }
  { GemmArgs g = mk(); g.seg[0] = sid(nb3, 128, 128); g.nseg = 1;
    g.W = fm2.w; g.ldw = 4; g.bias = fm2.b;
    g.res = states; g.ldres = 4;
    g.C = (float*)d_out; g.ldc = 4;
    g.M = NN; g.N = 4; g.K = 128; launch_gemm(stream, g); }
}
